// EdgeClassifier_29910152249703
// MI455X (gfx1250) — compile-verified, matrix operand order checked by reading, not run on hardware
//
#include <hip/hip_runtime.h>

#define NN 100000   // nodes
#define NE 1000000  // edges
#define NF 128      // node features
#define HID 64

typedef unsigned short u16;
typedef __attribute__((ext_vector_type(4)))  float  f32x4;
typedef __attribute__((ext_vector_type(8)))  float  v8f;
typedef __attribute__((ext_vector_type(8)))  u16    u16x8;
typedef __attribute__((ext_vector_type(16))) __bf16 v16bf;

__device__ __forceinline__ u16 f2bf(float f) {
  unsigned x = __float_as_uint(f);
  unsigned r = (x + 0x7FFFu + ((x >> 16) & 1u)) >> 16;  // RNE
  return (u16)r;
}

__device__ __forceinline__ u16x8 cvt8(f32x4 a, f32x4 b) {
  u16x8 r;
  r[0]=f2bf(a[0]); r[1]=f2bf(a[1]); r[2]=f2bf(a[2]); r[3]=f2bf(a[3]);
  r[4]=f2bf(b[0]); r[5]=f2bf(b[1]); r[6]=f2bf(b[2]); r[7]=f2bf(b[3]);
  return r;
}

// 16x32 bf16 WMMA operand fragment (A row-major view / B via pre-transposed W).
// tile points at element (r=0,k=0); ldk = row stride (elements). 16B aligned.
// Lanes 0-15: r=lane, K halves {0..7,16..23}; lanes 16-31: r=lane-16, {8..15,24..31}.
__device__ __forceinline__ v16bf load_frag(const u16* tile, int ldk) {
  int lane = threadIdx.x & 31;
  int r  = lane & 15;
  int kb = (lane & 16) ? 8 : 0;
  const u16* p = tile + r * ldk + kb;
  union { struct { u16x8 lo, hi; } s; v16bf v; } u;
  u.s.lo = *(const u16x8*)(p);
  u.s.hi = *(const u16x8*)(p + 16);
  return u.v;
}

// ---- weight prep: f32 (K x N) row-major -> bf16 (N x Kpad), zero-padded ----
__global__ void k_transpose(const float* __restrict__ S, u16* __restrict__ D,
                            int K, int N, int Kpad) {
  int i = blockIdx.x * blockDim.x + threadIdx.x;
  if (i >= N * Kpad) return;
  int n = i / Kpad, k = i - n * Kpad;
  D[i] = (k < K) ? f2bf(S[(size_t)k * N + n]) : (u16)0;
}

__global__ void k_zero(float* __restrict__ p, long long n) {
  long long i = (long long)blockIdx.x * blockDim.x + threadIdx.x;
  if (i < n) p[i] = 0.f;
}

__global__ void k_degree(const int* __restrict__ dstI, float* __restrict__ deg, int nE) {
  int e = blockIdx.x * blockDim.x + threadIdx.x;
  if (e < nE) atomicAdd(&deg[dstI[e]], 1.0f);
}

// ---- C(Mx64,f32) = A(MxKin,f32) @ W (via Wt = W^T bf16, 64 x Kin) ----
// block: 256 thr (8 waves), 64-row tile; wave w -> rows 16*(w&3), cols 32*(w>>2)
__global__ void k_gemm(const float* __restrict__ A, int M, int Kin,
                       const u16* __restrict__ Wt, float* __restrict__ C) {
  __shared__ __align__(16) u16 As[64][40];  // 64 x 32 bf16, +8 pad (16B rows)
  int t = threadIdx.x;
  int wv = t >> 5, lane = t & 31;
  int rt = (wv & 3) * 16;
  int ct = (wv >> 2) * 32;
  int m0 = blockIdx.x * 64;
  v8f acc0 = {}, acc1 = {};
  for (int k0 = 0; k0 < Kin; k0 += 32) {
    int row = t >> 2, cg = (t & 3) * 8;
    int gr = m0 + row; if (gr > M - 1) gr = M - 1;
    const float* ap = A + (size_t)gr * Kin + k0 + cg;
    f32x4 fa = *(const f32x4*)ap;
    f32x4 fb = *(const f32x4*)(ap + 4);
    *(u16x8*)&As[row][cg] = cvt8(fa, fb);
    __syncthreads();
    v16bf a  = load_frag(&As[rt][0], 40);
    v16bf b0 = load_frag(Wt + (size_t)(ct)      * Kin + k0, Kin);
    v16bf b1 = load_frag(Wt + (size_t)(ct + 16) * Kin + k0, Kin);
    acc0 = __builtin_amdgcn_wmma_f32_16x16x32_bf16(false, a, false, b0, (short)0, acc0, false, false);
    acc1 = __builtin_amdgcn_wmma_f32_16x16x32_bf16(false, a, false, b1, (short)0, acc1, false, false);
    __syncthreads();
  }
  int n = lane & 15, mo = (lane & 16) ? 8 : 0;
#pragma unroll
  for (int i = 0; i < 8; i++) {
    int gm = m0 + rt + mo + i;
    if (gm < M) {
      C[(size_t)gm * HID + ct + n]      = acc0[i];
      C[(size_t)gm * HID + ct + 16 + n] = acc1[i];
    }
  }
}

// ---- agg[dst] += P[src], 64 lanes per edge ----
__global__ void k_scatter(const int* __restrict__ srcI, const int* __restrict__ dstI,
                          const float* __restrict__ P, float* __restrict__ agg, int nE) {
  long long idx = (long long)blockIdx.x * blockDim.x + threadIdx.x;
  int e = (int)(idx >> 6);
  if (e >= nE) return;
  int c = (int)(idx & 63);
  atomicAdd(&agg[(size_t)dstI[e] * HID + c], P[(size_t)srcI[e] * HID + c]);
}

// ---- H = relu(LN(agg/max(deg,1) + bl + R)), one wave per node ----
__global__ void k_node_update(const float* __restrict__ agg, const float* __restrict__ deg,
                              const float* __restrict__ R, const float* __restrict__ bl,
                              const float* __restrict__ g, const float* __restrict__ be,
                              float* __restrict__ H, int M) {
  int wid  = (blockIdx.x * blockDim.x + threadIdx.x) >> 5;
  int lane = threadIdx.x & 31;
  if (wid >= M) return;
  float inv = 1.0f / fmaxf(deg[wid], 1.0f);
  size_t b0 = (size_t)wid * HID;
  float v0 = agg[b0 + lane]      * inv + bl[lane]      + R[b0 + lane];
  float v1 = agg[b0 + lane + 32] * inv + bl[lane + 32] + R[b0 + lane + 32];
  float s = v0 + v1, ss = v0 * v0 + v1 * v1;
#pragma unroll
  for (int off = 16; off >= 1; off >>= 1) {
    s  += __shfl_xor(s,  off, 32);
    ss += __shfl_xor(ss, off, 32);
  }
  float mu  = s  * (1.0f / 64.0f);
  float var = ss * (1.0f / 64.0f) - mu * mu;
  float rs  = rsqrtf(var + 1e-5f);
  float o0 = (v0 - mu) * rs * g[lane]      + be[lane];
  float o1 = (v1 - mu) * rs * g[lane + 32] + be[lane + 32];
  H[b0 + lane]      = o0 > 0.f ? o0 : 0.f;
  H[b0 + lane + 32] = o1 > 0.f ? o1 : 0.f;
}

// ---- fused edge MLP: 16 edges per wave, WMMA for 160->64 and 64->32 ----
#define EMW 8          // waves per block
#define ESTR 5376      // per-wave LDS bytes (16 x 168 u16), reused per phase
__global__ void k_edge_mlp(const int* __restrict__ srcI, const int* __restrict__ dstI,
                           const float* __restrict__ H,  const float* __restrict__ EA,
                           const u16* __restrict__ W1t,  const float* __restrict__ b1,
                           const u16* __restrict__ W2t,  const float* __restrict__ b2,
                           const float* __restrict__ W3, const float* __restrict__ b3,
                           float* __restrict__ out, int nE) {
  __shared__ __align__(16) unsigned char smem[EMW * ESTR];
  int wv = threadIdx.x >> 5, lane = threadIdx.x & 31;
  int e0 = (blockIdx.x * EMW + wv) * 16;
  u16* At = (u16*)(smem + wv * ESTR);  // 16 x 168 bf16 (160 used)
  // stage [h_src | h_dst | ea | 0-pad] for 16 edges
  int m = lane & 15;
  int e = e0 + m; if (e > nE - 1) e = nE - 1;
  int node = (lane < 16) ? srcI[e] : dstI[e];
  const float* hp = H + (size_t)node * HID;
  int base = (lane < 16) ? 0 : 64;
#pragma unroll
  for (int j = 0; j < 8; j++) {
    f32x4 fa = *(const f32x4*)(hp + j * 8);
    f32x4 fb = *(const f32x4*)(hp + j * 8 + 4);
    *(u16x8*)&At[m * 168 + base + j * 8] = cvt8(fa, fb);
  }
  if (lane < 16) {
    const float* ep = EA + (size_t)e * 16;
    f32x4 a0 = *(const f32x4*)(ep);     f32x4 a1 = *(const f32x4*)(ep + 4);
    f32x4 a2 = *(const f32x4*)(ep + 8); f32x4 a3 = *(const f32x4*)(ep + 12);
    *(u16x8*)&At[m * 168 + 128] = cvt8(a0, a1);
    *(u16x8*)&At[m * 168 + 136] = cvt8(a2, a3);
    u16x8 z = {};
    *(u16x8*)&At[m * 168 + 144] = z;
    *(u16x8*)&At[m * 168 + 152] = z;
  }
  __syncthreads();
  // GEMM1: (16x160) @ (160x64)
  v8f acc[4] = {{}, {}, {}, {}};
#pragma unroll
  for (int kc = 0; kc < 5; kc++) {
    v16bf a = load_frag(At + kc * 32, 168);
#pragma unroll
    for (int nt = 0; nt < 4; nt++) {
      v16bf b = load_frag(W1t + (size_t)(nt * 16) * 160 + kc * 32, 160);
      acc[nt] = __builtin_amdgcn_wmma_f32_16x16x32_bf16(false, a, false, b, (short)0, acc[nt], false, false);
    }
  }
  __syncthreads();
  // bias + relu -> z1 bf16 (16 x 72), reusing LDS
  u16* Zt = (u16*)(smem + wv * ESTR);
  int ncol = lane & 15, mo = (lane & 16) ? 8 : 0;
#pragma unroll
  for (int nt = 0; nt < 4; nt++) {
    float bc = b1[nt * 16 + ncol];
#pragma unroll
    for (int i = 0; i < 8; i++) {
      float v = acc[nt][i] + bc; v = v > 0.f ? v : 0.f;
      Zt[(mo + i) * 72 + nt * 16 + ncol] = f2bf(v);
    }
  }
  __syncthreads();
  // GEMM2: (16x64) @ (64x32)
  v8f acc2[2] = {{}, {}};
#pragma unroll
  for (int kc = 0; kc < 2; kc++) {
    v16bf a = load_frag(Zt + kc * 32, 72);
#pragma unroll
    for (int nt = 0; nt < 2; nt++) {
      v16bf b = load_frag(W2t + (size_t)(nt * 16) * 64 + kc * 32, 64);
      acc2[nt] = __builtin_amdgcn_wmma_f32_16x16x32_bf16(false, a, false, b, (short)0, acc2[nt], false, false);
    }
  }
  __syncthreads();
  float* Z2 = (float*)(smem + wv * ESTR);  // 16 x 33 f32
#pragma unroll
  for (int nt = 0; nt < 2; nt++) {
    float bc = b2[nt * 16 + ncol];
#pragma unroll
    for (int i = 0; i < 8; i++) {
      float v = acc2[nt][i] + bc; v = v > 0.f ? v : 0.f;
      Z2[(mo + i) * 33 + nt * 16 + ncol] = v;
    }
  }
  __syncthreads();
  if (lane < 16) {
    int eo = e0 + lane;
    if (eo < nE) {
      float s = b3[0];
#pragma unroll
      for (int n2 = 0; n2 < 32; n2++) s += Z2[lane * 33 + n2] * W3[n2];
      out[eo] = s;
    }
  }
}

extern "C" void kernel_launch(void* const* d_in, const int* in_sizes, int n_in,
                              void* d_out, int out_size, void* d_ws, size_t ws_size,
                              hipStream_t stream) {
  (void)in_sizes; (void)n_in; (void)out_size; (void)ws_size;
  const float* x   = (const float*)d_in[0];
  const int*   ei  = (const int*)d_in[1];
  const float* ea  = (const float*)d_in[2];
  const float* Wl0 = (const float*)d_in[3];
  const float* bl0 = (const float*)d_in[4];
  const float* Wr0 = (const float*)d_in[5];
  const float* g0  = (const float*)d_in[6];
  const float* be0 = (const float*)d_in[7];
  const float* Wl1 = (const float*)d_in[8];
  const float* bl1 = (const float*)d_in[9];
  const float* Wr1 = (const float*)d_in[10];
  const float* g1  = (const float*)d_in[11];
  const float* be1 = (const float*)d_in[12];
  const float* W1  = (const float*)d_in[13];
  const float* b1  = (const float*)d_in[14];
  const float* W2  = (const float*)d_in[15];
  const float* b2  = (const float*)d_in[16];
  const float* W3  = (const float*)d_in[17];
  const float* b3  = (const float*)d_in[18];
  float* out = (float*)d_out;
  const int* srcI = ei;
  const int* dstI = ei + NE;

  unsigned char* ws = (unsigned char*)d_ws;
  size_t off = 0;
  auto carve = [&](size_t bytes) -> unsigned char* {
    unsigned char* p = ws + off;
    off += (bytes + 255) & ~(size_t)255;
    return p;
  };
  u16*   Wl0t = (u16*)carve(64 * 128 * 2);
  u16*   Wr0t = (u16*)carve(64 * 128 * 2);
  u16*   Wl1t = (u16*)carve(64 * 64 * 2);
  u16*   Wr1t = (u16*)carve(64 * 64 * 2);
  u16*   W1t  = (u16*)carve(64 * 160 * 2);   // padded 144 -> 160
  u16*   W2t  = (u16*)carve(32 * 64 * 2);
  float* deg  = (float*)carve((size_t)NN * 4);
  float* P    = (float*)carve((size_t)NN * HID * 4);
  float* R    = (float*)carve((size_t)NN * HID * 4);
  float* agg  = (float*)carve((size_t)NN * HID * 4);
  float* h0   = (float*)carve((size_t)NN * HID * 4);
  float* h1   = (float*)carve((size_t)NN * HID * 4);

  const int B = 256;
  // weight prep
  k_transpose<<<(64 * 128 + B - 1) / B, B, 0, stream>>>(Wl0, Wl0t, 128, 64, 128);
  k_transpose<<<(64 * 128 + B - 1) / B, B, 0, stream>>>(Wr0, Wr0t, 128, 64, 128);
  k_transpose<<<(64 * 64  + B - 1) / B, B, 0, stream>>>(Wl1, Wl1t, 64, 64, 64);
  k_transpose<<<(64 * 64  + B - 1) / B, B, 0, stream>>>(Wr1, Wr1t, 64, 64, 64);
  k_transpose<<<(64 * 160 + B - 1) / B, B, 0, stream>>>(W1,  W1t,  144, 64, 160);
  k_transpose<<<(32 * 64  + B - 1) / B, B, 0, stream>>>(W2,  W2t,  64, 32, 64);
  // degrees
  k_zero<<<(NN + B - 1) / B, B, 0, stream>>>(deg, NN);
  k_degree<<<(NE + B - 1) / B, B, 0, stream>>>(dstI, deg, NE);

  const int gemmGrid = (NN + 63) / 64;
  const long long aggN = (long long)NN * HID;
  // ---- layer 0 ----
  k_gemm<<<gemmGrid, B, 0, stream>>>(x, NN, NF, Wl0t, P);   // P = x @ Wl0
  k_gemm<<<gemmGrid, B, 0, stream>>>(x, NN, NF, Wr0t, R);   // R = x @ Wr0
  k_zero<<<(int)((aggN + B - 1) / B), B, 0, stream>>>(agg, aggN);
  k_scatter<<<(int)(((long long)NE * 64 + B - 1) / B), B, 0, stream>>>(srcI, dstI, P, agg, NE);
  k_node_update<<<(NN * 32 + B - 1) / B, B, 0, stream>>>(agg, deg, R, bl0, g0, be0, h0, NN);
  // ---- layer 1 ----
  k_gemm<<<gemmGrid, B, 0, stream>>>(h0, NN, HID, Wl1t, P);
  k_gemm<<<gemmGrid, B, 0, stream>>>(h0, NN, HID, Wr1t, R);
  k_zero<<<(int)((aggN + B - 1) / B), B, 0, stream>>>(agg, aggN);
  k_scatter<<<(int)(((long long)NE * 64 + B - 1) / B), B, 0, stream>>>(srcI, dstI, P, agg, NE);
  k_node_update<<<(NN * 32 + B - 1) / B, B, 0, stream>>>(agg, deg, R, bl1, g1, be1, h1, NN);
  // ---- fused edge MLP ----
  k_edge_mlp<<<(NE + EMW * 16 - 1) / (EMW * 16), B, 0, stream>>>(
      srcI, dstI, h1, ea, W1t, b1, W2t, b2, W3, b3, out, NE);
}
